// LTC_NCP_RNN_88768384074313
// MI455X (gfx1250) — compile-verified, matrix operand order checked by reading, not run on hardware
//
#include <hip/hip_runtime.h>
#include <hip/hip_bf16.h>
#include <math.h>

// ---------------------------------------------------------------------------
// LTC (liquid time-constant) 2-level RNN for MI455X (gfx1250, wave32, WMMA).
//
// Shapes: B=64, T=128, E=256, HH=256, V=50000. out = [B, T, 2*HH] f32.
//
//   0) pack_bf16_kernel  : one-shot f32->bf16 pack of W_in (kills redundant
//      per-tile v_cvt work in the GEMM; B-frags become raw b128 loads).
//   1) xproj_gemm_kernel : xp[t,b,n] = sum_d X[r,d]*W[n,d] + bias[n]
//      (scan-major [T][B][H] layout so each step's drive is contiguous 64KB)
//   2) ltc_scan_kernel   : persistent single-workgroup (1024 thr = 32 waves)
//      sequential scan; Wr = W_rec*mask pre-packed to bf16 B-fragments in
//      registers; h in f32 registers + bf16 LDS; xp C-init double-buffered
//      in registers + global_prefetch of step t+2.
// ---------------------------------------------------------------------------

typedef __attribute__((ext_vector_type(16))) __bf16 v16bf;
typedef __attribute__((ext_vector_type(8)))  __bf16 v8bf;
typedef __attribute__((ext_vector_type(8)))  float  v8f;

#define LTC_B  64
#define LTC_T  128
#define LTC_H  256   // HH (per-level hidden)
#define LTC_E  256

union FragU { v16bf v; v8bf h[2]; };

// D = A*B + C, bf16 inputs, f32 accumulate (v_wmma_f32_16x16x32_bf16)
__device__ __forceinline__ v8f wmma_bf16(v16bf a, v16bf b, v8f c) {
  return __builtin_amdgcn_wmma_f32_16x16x32_bf16(
      /*neg_a=*/false, a, /*neg_b=*/false, b,
      /*c_mod=*/(short)0, c, /*reuse_a=*/false, /*reuse_b=*/false);
}

// Build a v16bf fragment from two contiguous 8-float runs (f32 -> bf16).
__device__ __forceinline__ v16bf load_frag_f32(const float* __restrict__ p0,
                                               const float* __restrict__ p1) {
  v16bf r;
#pragma unroll
  for (int i = 0; i < 8; ++i) {
    r[i]     = (__bf16)p0[i];
    r[i + 8] = (__bf16)p1[i];
  }
  return r;
}

// Fragment of Wr = W_rec * mask (elementwise), packed to bf16.
__device__ __forceinline__ v16bf load_frag_mul(const float* __restrict__ w0,
                                               const float* __restrict__ m0,
                                               const float* __restrict__ w1,
                                               const float* __restrict__ m1) {
  v16bf r;
#pragma unroll
  for (int i = 0; i < 8; ++i) {
    r[i]     = (__bf16)(w0[i] * m0[i]);
    r[i + 8] = (__bf16)(w1[i] * m1[i]);
  }
  return r;
}

// Fragment from bf16 memory (two aligned 16-byte runs -> *_load_b128 x2).
__device__ __forceinline__ v16bf load_frag_bf16(const __bf16* p0,
                                                const __bf16* p1) {
  FragU u;
  u.h[0] = *(const v8bf*)p0;
  u.h[1] = *(const v8bf*)p1;
  return u.v;
}

// ---------------------------------------------------------------------------
// Kernel 0: one-shot f32 -> bf16 pack (row-major preserved).
// ---------------------------------------------------------------------------
__global__ __launch_bounds__(256)
void pack_bf16_kernel(__bf16* __restrict__ dst, const float* __restrict__ src,
                      int n) {
  const int i = (blockIdx.x * 256 + threadIdx.x) * 8;
  if (i + 8 <= n) {
    const v8f v = *(const v8f*)(src + i);
    v8bf o;
#pragma unroll
    for (int j = 0; j < 8; ++j) o[j] = (__bf16)v[j];
    *(v8bf*)(dst + i) = o;
  }
}

// ---------------------------------------------------------------------------
// Kernel 1: input projection GEMM (optionally fused embedding gather).
//   xp[t, b, n] = sum_d X[r, d] * W[n, d] + bias[n],  r = b*T + t
// 256 threads (8 waves); each wave does a 16x64 strip: per K-step one
// A-fragment (f32->bf16 cvt) reused across 4 N-tiles whose B-fragments are
// raw bf16 b128 loads from the pre-packed weight buffer.
// ---------------------------------------------------------------------------
__global__ __launch_bounds__(256)
void xproj_gemm_kernel(float* __restrict__ xp,         // [T][B][H] scan-major
                       const float* __restrict__ src,  // emb or out rows
                       const int* __restrict__ tokens, // null => direct rows
                       int src_stride,                 // 256 (emb) / 512 (out)
                       const __bf16* __restrict__ Wbf, // [256,256] bf16 packed
                       const float* __restrict__ bias) // [256]
{
  const int tid  = threadIdx.x;
  const int wv   = tid >> 5;
  const int lane = tid & 31;
  const int l16  = lane & 15;
  const int g    = lane >> 4;

  const int job    = blockIdx.x * 8 + wv;  // 0..2047
  const int mtile  = job >> 2;             // 0..511 (16 rows each)
  const int nstrip = job & 3;              // 0..3   (64 cols each)

  // A row pointer for this lane's M = l16.
  const int rowA = mtile * 16 + l16;
  const float* __restrict__ arow =
      tokens ? (src + (size_t)tokens[rowA] * src_stride)
             : (src + (size_t)rowA * src_stride);

  v8f acc[4] = {};

#pragma unroll
  for (int kk = 0; kk < 8; ++kk) {
    const int k0 = kk * 32 + g * 8;
    const v16bf a = load_frag_f32(arow + k0, arow + k0 + 16);
#pragma unroll
    for (int nn = 0; nn < 4; ++nn) {
      const int col = nstrip * 64 + nn * 16 + l16;
      const __bf16* __restrict__ wrow = Wbf + (size_t)col * LTC_E + k0;
      const v16bf b = load_frag_bf16(wrow, wrow + 16);
      acc[nn] = wmma_bf16(a, b, acc[nn]);
    }
  }

#pragma unroll
  for (int nn = 0; nn < 4; ++nn) {
    const int col = nstrip * 64 + nn * 16 + l16;
    const float bs = bias[col];
#pragma unroll
    for (int r = 0; r < 8; ++r) {
      const int row = mtile * 16 + r + 8 * g;  // C/D layout: M = r + 8*(lane/16)
      const int b   = row >> 7;                // batch index
      const int t   = row & 127;               // time index
      xp[((size_t)t * LTC_B + b) * LTC_H + col] = acc[nn][r] + bs;
    }
  }
}

// ---------------------------------------------------------------------------
// Kernel 2: persistent sequential LTC scan, one workgroup (32 waves).
//   pre    = xp[t] + h @ (W_rec*mask)^T
//   h_new  = h + (tanh(pre) - h) / (softplus(tau_raw) + 0.1)     (DT = 1)
// Tile map: wave w -> N-tile nt = w&15, M-tiles mt0 = w>>4 and mt0+2.
// Both tiles share one set of 8 register-resident B fragments (64 VGPRs).
// xp C-inits are software-pipelined (consumed one step after their loads).
// ---------------------------------------------------------------------------
__global__ __launch_bounds__(1024)
void ltc_scan_kernel(float* __restrict__ out,       // [B, T, 512]
                     const float* __restrict__ xp,  // [T, B, 256]
                     const float* __restrict__ Wrec,
                     const float* __restrict__ Mask,
                     const float* __restrict__ tau_raw,
                     int lvl)
{
  __shared__ __align__(16) __bf16 h_lds[LTC_B][LTC_H];  // 32 KB

  const int tid  = threadIdx.x;
  const int wave = tid >> 5;        // 0..31
  const int lane = tid & 31;
  const int l16  = lane & 15;
  const int g    = lane >> 4;

  const int nt   = wave & 15;       // N-tile
  const int mt0  = wave >> 4;       // 0 or 1; second tile at mt0+2
  const int col  = nt * 16 + l16;
  const int row0 = mt0 * 16;
  const int row1 = (mt0 + 2) * 16;

  // Per-column time constant: tau = softplus(tau_raw) + 0.1
  const float inv_tau = 1.0f / (log1pf(expf(tau_raw[col])) + 0.1f);

  // Preload B fragments: B[k, n] = Wrec[n, k] * Mask[n, k]  (i.e. Wr^T).
  v16bf Bf[8];
#pragma unroll
  for (int kk = 0; kk < 8; ++kk) {
    const int k0 = kk * 32 + g * 8;
    const float* __restrict__ w0 = Wrec + (size_t)col * LTC_H + k0;
    const float* __restrict__ m0 = Mask + (size_t)col * LTC_H + k0;
    Bf[kk] = load_frag_mul(w0, m0, w0 + 16, m0 + 16);
  }

  // h = 0 in LDS (bf16) and registers (f32).
  {
    v8bf z;
#pragma unroll
    for (int j = 0; j < 8; ++j) z[j] = (__bf16)0.0f;
    v8bf* p = (v8bf*)&h_lds[0][0];
    for (int i = tid; i < (LTC_B * LTC_H) / 8; i += 1024) p[i] = z;
  }
  v8f h0v = {};
  v8f h1v = {};

  // Software-pipelined xp loads: step t consumes values fetched at step t-1.
  v8f xc0, xc1;
#pragma unroll
  for (int r = 0; r < 8; ++r) {
    xc0[r] = xp[((size_t)0 * LTC_B + row0 + r + 8 * g) * LTC_H + col];
    xc1[r] = xp[((size_t)0 * LTC_B + row1 + r + 8 * g) * LTC_H + col];
  }
  __syncthreads();

  for (int t = 0; t < LTC_T; ++t) {
    // C init = input drive xp[t, b, col] (already in registers).
    v8f acc0 = xc0;
    v8f acc1 = xc1;

    // Issue next step's xp loads now; they retire under the WMMA section.
    const int tn = (t + 1 < LTC_T) ? (t + 1) : t;
#pragma unroll
    for (int r = 0; r < 8; ++r) {
      xc0[r] = xp[((size_t)tn * LTC_B + row0 + r + 8 * g) * LTC_H + col];
      xc1[r] = xp[((size_t)tn * LTC_B + row1 + r + 8 * g) * LTC_H + col];
    }
    // Pull step t+2's contiguous 64KB drive toward this WGP (1 line/thread).
    const int tp = (t + 2 < LTC_T) ? (t + 2) : t;
    __builtin_prefetch(xp + (size_t)tp * LTC_B * LTC_H + tid * 16, 0, 0);

    // pre += h @ Wr^T : 8 K-steps x 2 tiles = 16 WMMAs per wave per step.
#pragma unroll
    for (int kk = 0; kk < 8; ++kk) {
      const int k0 = kk * 32 + g * 8;
      const v16bf a0 = load_frag_bf16(&h_lds[row0 + l16][k0],
                                      &h_lds[row0 + l16][k0 + 16]);
      acc0 = wmma_bf16(a0, Bf[kk], acc0);
      const v16bf a1 = load_frag_bf16(&h_lds[row1 + l16][k0],
                                      &h_lds[row1 + l16][k0 + 16]);
      acc1 = wmma_bf16(a1, Bf[kk], acc1);
    }
    __syncthreads();  // all waves done reading h_lds(t)

    // Elementwise LTC update; publish h(t+1) to LDS (bf16) and out (f32).
#pragma unroll
    for (int r = 0; r < 8; ++r) {
      const int b0 = row0 + r + 8 * g;
      const int b1 = row1 + r + 8 * g;
      const float hn0 = h0v[r] + (tanhf(acc0[r]) - h0v[r]) * inv_tau;
      const float hn1 = h1v[r] + (tanhf(acc1[r]) - h1v[r]) * inv_tau;
      h0v[r] = hn0;
      h1v[r] = hn1;
      out[((size_t)b0 * LTC_T + t) * 512 + (size_t)lvl * 256 + col] = hn0;
      out[((size_t)b1 * LTC_T + t) * 512 + (size_t)lvl * 256 + col] = hn1;
      h_lds[b0][col] = (__bf16)hn0;
      h_lds[b1][col] = (__bf16)hn1;
    }
    __syncthreads();  // h_lds(t+1) visible before next step's reads
  }
}

// ---------------------------------------------------------------------------
// Workspace layout: [0, 8MB) xp (scan-major), [8MB, 8MB+128KB) packed W bf16.
// ---------------------------------------------------------------------------
extern "C" void kernel_launch(void* const* d_in, const int* in_sizes, int n_in,
                              void* d_out, int out_size, void* d_ws,
                              size_t ws_size, hipStream_t stream) {
  const int*   tokens = (const int*)  d_in[0];
  const float* emb    = (const float*)d_in[1];
  const float* W_in0  = (const float*)d_in[2];
  const float* W_rec0 = (const float*)d_in[3];
  const float* b0     = (const float*)d_in[4];
  const float* tau0   = (const float*)d_in[5];
  const float* mask0  = (const float*)d_in[6];
  const float* W_in1  = (const float*)d_in[7];
  const float* W_rec1 = (const float*)d_in[8];
  const float* b1     = (const float*)d_in[9];
  const float* tau1   = (const float*)d_in[10];
  const float* mask1  = (const float*)d_in[11];

  float*  out = (float*)d_out;
  float*  xp  = (float*)d_ws;                                // 8 MB
  __bf16* Wbf = (__bf16*)((char*)d_ws + ((size_t)8 << 20));  // 128 KB

  (void)in_sizes; (void)n_in; (void)out_size; (void)ws_size;

  const int nW = LTC_H * LTC_E;  // 65536 weight elements

  // Level 0: pack W_in0, fused embedding-gather GEMM, sequential scan.
  pack_bf16_kernel<<<nW / (256 * 8), 256, 0, stream>>>(Wbf, W_in0, nW);
  xproj_gemm_kernel<<<256, 256, 0, stream>>>(xp, emb, tokens, LTC_E, Wbf, b0);
  ltc_scan_kernel<<<1, 1024, 0, stream>>>(out, xp, W_rec0, mask0, tau0, 0);

  // Level 1: pack W_in1, GEMM over h0_seq rows (stride 512 in out), scan.
  pack_bf16_kernel<<<nW / (256 * 8), 256, 0, stream>>>(Wbf, W_in1, nW);
  xproj_gemm_kernel<<<256, 256, 0, stream>>>(xp, out, nullptr, 512, Wbf, b1);
  ltc_scan_kernel<<<1, 1024, 0, stream>>>(out, xp, W_rec1, mask1, tau1, 1);
}
